// OSNet_43997644980535
// MI455X (gfx1250) — matrix operand order repaired and checked by reading, not compiled, not run on hardware
//
#include <hip/hip_runtime.h>
#include <stdint.h>
#include <stddef.h>

// ---------- problem constants (from setup_inputs) ----------
#define BATCH 8
#define NNODE 1024
#define DIN   2048
#define DOUT  512
#define PBLK  8
#define TWIN  4
#define NB    (NNODE / PBLK)   // 128 blocks
#define HCH   (2 * DOUT)       // 1024 channels of h
#define ROWS  (BATCH * NNODE)  // 8192 rows
#define KC    64               // GEMM k-chunk staged in LDS

typedef __attribute__((ext_vector_type(16))) __bf16        bf16x16;
typedef __attribute__((ext_vector_type(8)))  float         f32x8;
typedef __attribute__((ext_vector_type(4)))  unsigned int  u32x4;
typedef __attribute__((ext_vector_type(8)))  int           i32x8;
typedef __attribute__((ext_vector_type(4)))  int           i32x4;

union Frag { bf16x16 v; u32x4 q[2]; };

__device__ __forceinline__ unsigned short f2bf(float f) {
    unsigned int u = __float_as_uint(f);
    u += 0x7FFFu + ((u >> 16) & 1u);           // round-to-nearest-even
    return (unsigned short)(u >> 16);
}

// ---- Tensor Data Mover: 2-D tile (tile_k x tile_rows, 2-byte elems) ----
// Descriptor packing per CDNA5 ISA.
// This toolchain exposes the 6-arg builtin (extra int32x8 group before cpol).
__device__ __forceinline__ void tdm_load_2d(const unsigned short* gptr,
                                            unsigned lds_byte_addr,
                                            int tile_k, int tile_rows,
                                            unsigned long long row_stride_elems) {
    unsigned long long ga = (unsigned long long)(size_t)gptr;   // byte address
    u32x4 g0;
    g0.x = 1u;                                                  // count=1, user desc
    g0.y = lds_byte_addr;                                       // lds_addr [63:32]
    g0.z = (unsigned)(ga & 0xFFFFFFFFu);                        // global_addr lo
    g0.w = (unsigned)((ga >> 32) & 0x01FFFFFFu) | (2u << 30);   // addr hi | type=2

    unsigned long long dim0 = 0x40000000ull;    // huge tensor dims: no OOB clip
    unsigned long long dim1 = 0x40000000ull;
    unsigned long long s0   = row_stride_elems; // tensor_dim0_stride (48-bit)
    // group1 bits [63:0]
    unsigned long long q0 = (1ull << 16)                         // data_size=2B
                          | ((dim0 & 0xFFFFull) << 48);          // tensor_dim0[15:0]
    // bits [127:64]
    unsigned long long q1 = ((dim0 >> 16) & 0xFFFFull)           // tensor_dim0[31:16]
                          | ((dim1 & 0xFFFFFFFFull) << 16)       // tensor_dim1
                          | (((unsigned long long)(tile_k & 0xFFFF)) << 48); // tile_dim0
    // bits [191:128]
    unsigned long long q2 = ((unsigned long long)(tile_rows & 0xFFFF))       // tile_dim1
                          | ((s0 & 0xFFFFFFFFull) << 32);        // stride0[31:0]
    // bits [255:192]
    unsigned long long q3 = ((s0 >> 32) & 0xFFFFull);            // stride0[47:32]

    i32x8 g1;
    g1[0] = (int)q0; g1[1] = (int)(q0 >> 32);
    g1[2] = (int)q1; g1[3] = (int)(q1 >> 32);
    g1[4] = (int)q2; g1[5] = (int)(q2 >> 32);
    g1[6] = (int)q3; g1[7] = (int)(q3 >> 32);
    i32x4 z4 = {0, 0, 0, 0};                    // 2-D tensor: groups 2/3 unused
    i32x8 z8 = {0, 0, 0, 0, 0, 0, 0, 0};
    __builtin_amdgcn_tensor_load_to_lds(g0, g1, z4, z4, z8, 0);
}

__device__ __forceinline__ unsigned lds_off(const void* p) {
    return (unsigned)(size_t)p;                 // flat LDS addr truncates to offset
}

// ---------------- elementwise fp32 -> bf16 ----------------
__global__ void cvt_bf16_kernel(const float* __restrict__ src,
                                unsigned short* __restrict__ dst, int n) {
    int i = blockIdx.x * blockDim.x + threadIdx.x;
    int stride = gridDim.x * blockDim.x;
    for (; i < n; i += stride) dst[i] = f2bf(src[i]);
}

// ---------------- per-row inverse L2 norm -----------------
__global__ void row_inv_norm_kernel(const float* __restrict__ src,
                                    float* __restrict__ dst, int cols) {
    __shared__ float red[256];
    int row = blockIdx.x;
    const float* p = src + (size_t)row * cols;
    float ss = 0.f;
    for (int c = threadIdx.x; c < cols; c += 256) { float v = p[c]; ss += v * v; }
    red[threadIdx.x] = ss;
    __syncthreads();
    for (int st = 128; st > 0; st >>= 1) {
        if (threadIdx.x < st) red[threadIdx.x] += red[threadIdx.x + st];
        __syncthreads();
    }
    if (threadIdx.x == 0) dst[row] = 1.0f / fmaxf(sqrtf(red[0]), 1e-12f);
}

// -------- block-local similarity (WMMA) + top-3 select --------
__global__ void sim_topk_kernel(const unsigned short* __restrict__ xbf,
                                const float* __restrict__ rinv,
                                int* __restrict__ nidx) {
    __shared__ float simb[8][PBLK][80];          // [wave][q][w], 20.5 KB
    int lane = threadIdx.x & 31;
    int wid  = threadIdx.x >> 5;
    int job  = blockIdx.x * 8 + wid;             // 0 .. BATCH*NB-1 (exact)
    int batch = job >> 7;                        // / NB
    int blk   = job & (NB - 1);
    int sb = blk - TWIN; if (sb < 0) sb = 0;
    int eb = blk + TWIN + 1; if (eb > NB) eb = NB;
    int s = sb * PBLK, W = eb * PBLK - s;        // 40 <= W <= 72

    const unsigned short* xb = xbf + (size_t)batch * NNODE * DIN;

    f32x8 zero8 = {0.f,0.f,0.f,0.f,0.f,0.f,0.f,0.f};
    f32x8 acc[5];
#pragma unroll
    for (int t = 0; t < 5; ++t) acc[t] = zero8;

    int qm = lane & 15;
    bool azero = (lane & 8) != 0;                // pad rows 8..15 of A with 0
    u32x4 zq = {0u, 0u, 0u, 0u};
    const unsigned short* arow = xb + (size_t)(blk * PBLK + (qm & 7)) * DIN;

    for (int k0 = 0; k0 < DIN; k0 += 32) {
        Frag a;
        int kof = k0 + ((lane & 16) ? 16 : 0);
        const u32x4* ap = (const u32x4*)(arow + kof);
        a.q[0] = ap[0];
        a.q[1] = ap[1];
        if (azero) { a.q[0] = zq; a.q[1] = zq; }

        int kb = k0 + ((lane & 16) ? 16 : 0);
#pragma unroll
        for (int tl = 0; tl < 5; ++tl) {
            int w  = tl * 16 + (lane & 15);
            int wc = (w < W - 1) ? w : (W - 1);  // clamp like cand[]
            const u32x4* bp = (const u32x4*)(xb + (size_t)(s + wc) * DIN + kb);
            Frag b; b.q[0] = bp[0]; b.q[1] = bp[1];
            acc[tl] = __builtin_amdgcn_wmma_f32_16x16x32_bf16(
                false, a.v, false, b.v, (short)0, acc[tl], false, false);
        }
    }

    if (lane < 16) {
#pragma unroll
        for (int tl = 0; tl < 5; ++tl)
#pragma unroll
            for (int r = 0; r < 8; ++r)
                simb[wid][r][tl * 16 + lane] = acc[tl][r];
    }
    __syncthreads();

    if (lane < PBLK) {
        int q = lane;
        int grow = batch * NNODE + blk * PBLK + q;
        float rq = rinv[grow];
        float v1 = -1e30f, v2 = -1e30f, v3 = -1e30f;
        int   i1 = 0,      i2 = 0,      i3 = 0;
        for (int w = 0; w < W; ++w) {
            float sv = simb[wid][q][w] * rq * rinv[batch * NNODE + s + w];
            if (sv > v1)      { v3=v2; i3=i2; v2=v1; i2=i1; v1=sv; i1=w; }
            else if (sv > v2) { v3=v2; i3=i2; v2=sv; i2=w; }
            else if (sv > v3) { v3=sv; i3=w; }
        }
        (void)v1; (void)i1;                      // k=0 is self; keep k=1,2
        nidx[2 * grow + 0] = s + i2;
        nidx[2 * grow + 1] = s + i3;
    }
}

// -------- x_neib row = mean of 2 neighbor rows, stored bf16 --------
__global__ void gather_mean_bf16_kernel(const float* __restrict__ x,
                                        const int* __restrict__ nidx,
                                        unsigned short* __restrict__ dst) {
    int row   = blockIdx.x;
    int batch = row >> 10;
    int g1 = nidx[2 * row + 0];
    int g2 = nidx[2 * row + 1];
    const float* r1 = x + ((size_t)batch * NNODE + g1) * DIN;
    const float* r2 = x + ((size_t)batch * NNODE + g2) * DIN;
    unsigned short* o = dst + (size_t)row * DIN;
    for (int d = threadIdx.x; d < DIN; d += 256)
        o[d] = f2bf(0.5f * (r1[d] + r2[d]));
}

// -------- C[M,N] = A[M,K] * W[N,K]^T + bias, bf16 WMMA --------
// 8 waves/WG, WG tile 64x64. TDM stages A/B k-chunks into double-buffered
// LDS (TENSORcnt), waves read fragments with ds_load_b128, 4 WMMAs/chunk.
__global__ void gemm_bf16_kernel(const unsigned short* __restrict__ A,
                                 const unsigned short* __restrict__ Wt,
                                 const float* __restrict__ bias,
                                 float* __restrict__ out,
                                 int M, int K, int ldo, int ocol) {
    __shared__ unsigned short shA[2][64][KC];    // 2 x 8 KB
    __shared__ unsigned short shB[2][64][KC];    // 2 x 8 KB
    int lane = threadIdx.x & 31;
    int wid  = threadIdx.x >> 5;
    int mtiles = M >> 6;
    int bm = blockIdx.x % mtiles;
    int bn = blockIdx.x / mtiles;

    const unsigned short* Abase = A  + (size_t)(bm * 64) * K;
    const unsigned short* Bbase = Wt + (size_t)(bn * 64) * K;

    f32x8 c0 = {0.f,0.f,0.f,0.f,0.f,0.f,0.f,0.f};
    f32x8 c1 = c0;

    if (wid == 0) {                              // prologue: chunk 0 -> buf 0
        tdm_load_2d(Abase, lds_off(&shA[0][0][0]), KC, 64, (unsigned long long)K);
        tdm_load_2d(Bbase, lds_off(&shB[0][0][0]), KC, 64, (unsigned long long)K);
    }

    int mL  = (wid & 3) * 16 + (lane & 15);      // A row in LDS tile
    int nL0 = (wid >> 2) * 32 + (lane & 15);     // B row in LDS tile

    for (int kc = 0; kc < K; kc += KC) {
        int cur = (kc / KC) & 1;
        if (wid == 0) {
            if (kc + KC < K) {                   // prefetch next chunk -> other buf
                tdm_load_2d(Abase + (kc + KC), lds_off(&shA[cur ^ 1][0][0]),
                            KC, 64, (unsigned long long)K);
                tdm_load_2d(Bbase + (kc + KC), lds_off(&shB[cur ^ 1][0][0]),
                            KC, 64, (unsigned long long)K);
                __builtin_amdgcn_s_wait_tensorcnt(2);   // current chunk done
            } else {
                __builtin_amdgcn_s_wait_tensorcnt(0);
            }
        }
        __syncthreads();                         // chunk `cur` visible to all waves

        const unsigned short* aR  = &shA[cur][mL][0];
        const unsigned short* bR0 = &shB[cur][nL0][0];
        const unsigned short* bR1 = &shB[cur][nL0 + 16][0];
#pragma unroll
        for (int ks = 0; ks < KC; ks += 32) {
            Frag a, b0, b1;
            int kof = ks + ((lane & 16) ? 16 : 0);
            const u32x4* ap = (const u32x4*)(aR + kof);
            a.q[0] = ap[0]; a.q[1] = ap[1];
            int kb = ks + ((lane & 16) ? 16 : 0);
            const u32x4* bp0 = (const u32x4*)(bR0 + kb);
            b0.q[0] = bp0[0]; b0.q[1] = bp0[1];
            const u32x4* bp1 = (const u32x4*)(bR1 + kb);
            b1.q[0] = bp1[0]; b1.q[1] = bp1[1];
            c0 = __builtin_amdgcn_wmma_f32_16x16x32_bf16(false, a.v, false, b0.v,
                                                         (short)0, c0, false, false);
            c1 = __builtin_amdgcn_wmma_f32_16x16x32_bf16(false, a.v, false, b1.v,
                                                         (short)0, c1, false, false);
        }
        __syncthreads();                         // done reading before TDM overwrite
    }

    int tileM = bm * 64 + (wid & 3) * 16;
    int col   = bn * 64 + (wid >> 2) * 32 + (lane & 15);
    float bi0 = bias[col];
    float bi1 = bias[col + 16];
    int rbase = tileM + ((lane & 16) ? 8 : 0);
#pragma unroll
    for (int r = 0; r < 8; ++r) {
        size_t o = (size_t)(rbase + r) * ldo + ocol;
        out[o + col]      = c0[r] + bi0;
        out[o + col + 16] = c1[r] + bi1;
    }
}

// -------- zero scratch --------
__global__ void zero_kernel(float* __restrict__ p, int n) {
    for (int i = blockIdx.x * blockDim.x + threadIdx.x; i < n;
         i += gridDim.x * blockDim.x) p[i] = 0.f;
}

// -------- in-place row-normalize + ReLU + channel sum/sumsq --------
__global__ void bn_stats_kernel(float* __restrict__ h,
                                const float* __restrict__ hinv,
                                float* __restrict__ sums,
                                float* __restrict__ ssums) {
    int c0   = threadIdx.x * 4;
    int row0 = blockIdx.x * 32;
    float s[4] = {0,0,0,0}, q[4] = {0,0,0,0};
    for (int r = 0; r < 32; ++r) {
        int row = row0 + r;
        float inv = hinv[row];
        float* p = h + (size_t)row * HCH + c0;
#pragma unroll
        for (int j = 0; j < 4; ++j) {
            float v = fmaxf(p[j] * inv, 0.f);
            p[j] = v;
            s[j] += v; q[j] += v * v;
        }
    }
#pragma unroll
    for (int j = 0; j < 4; ++j) {
        atomicAdd(&sums[c0 + j],  s[j]);
        atomicAdd(&ssums[c0 + j], q[j]);
    }
}

// -------- in-place batch-norm apply --------
__global__ void bn_apply_kernel(float* __restrict__ h,
                                const float* __restrict__ sums,
                                const float* __restrict__ ssums,
                                const float* __restrict__ gamma,
                                const float* __restrict__ beta) {
    const float invn = 1.0f / (float)ROWS;
    int total = ROWS * HCH;
    int stride = gridDim.x * blockDim.x;
    for (int i = blockIdx.x * blockDim.x + threadIdx.x; i < total; i += stride) {
        int c = i & (HCH - 1);
        float mu  = sums[c] * invn;
        float var = ssums[c] * invn - mu * mu;
        h[i] = gamma[c] * (h[i] - mu) * rsqrtf(var + 1e-5f) + beta[c];
    }
}

extern "C" void kernel_launch(void* const* d_in, const int* in_sizes, int n_in,
                              void* d_out, int out_size, void* d_ws, size_t ws_size,
                              hipStream_t stream) {
    (void)in_sizes; (void)n_in; (void)out_size; (void)ws_size;
    const float* x     = (const float*)d_in[0];
    const float* Wx    = (const float*)d_in[1];
    const float* bx    = (const float*)d_in[2];
    const float* Wn    = (const float*)d_in[3];
    const float* bnv   = (const float*)d_in[4];
    const float* gamma = (const float*)d_in[5];
    const float* beta  = (const float*)d_in[6];
    float* out = (float*)d_out;                  // h lives here in place

    char* ws = (char*)d_ws;
    float* rinv  = (float*)(ws + 0);             //  8192 f32
    float* hinv  = (float*)(ws + 32768);         //  8192 f32
    float* sums  = (float*)(ws + 65536);         //  1024 f32
    float* ssums = (float*)(ws + 69632);         //  1024 f32 (contiguous w/ sums)
    int*   nidx  = (int*)  (ws + 73728);         //  8192*2 i32
    unsigned short* xbf  = (unsigned short*)(ws + 139264);            // 32 MB
    unsigned short* wxbf = xbf  + (size_t)ROWS * DIN;                 //  2 MB
    unsigned short* wnbf = wxbf + (size_t)DOUT * DIN;                 //  2 MB
    unsigned short* xnb  = wnbf + (size_t)DOUT * DIN;                 // 32 MB

    zero_kernel<<<1, 256, 0, stream>>>(sums, 2048);                   // sums+ssums
    cvt_bf16_kernel<<<4096, 256, 0, stream>>>(x,  xbf,  ROWS * DIN);
    cvt_bf16_kernel<<<512,  256, 0, stream>>>(Wx, wxbf, DOUT * DIN);
    cvt_bf16_kernel<<<512,  256, 0, stream>>>(Wn, wnbf, DOUT * DIN);
    row_inv_norm_kernel<<<ROWS, 256, 0, stream>>>(x, rinv, DIN);
    sim_topk_kernel<<<(BATCH * NB) / 8, 256, 0, stream>>>(xbf, rinv, nidx);
    gather_mean_bf16_kernel<<<ROWS, 256, 0, stream>>>(x, nidx, xnb);
    gemm_bf16_kernel<<<(ROWS / 64) * (DOUT / 64), 256, 0, stream>>>(
        xbf, wxbf, bx,  out, ROWS, DIN, HCH, 0);
    gemm_bf16_kernel<<<(ROWS / 64) * (DOUT / 64), 256, 0, stream>>>(
        xnb, wnbf, bnv, out, ROWS, DIN, HCH, DOUT);
    row_inv_norm_kernel<<<ROWS, 256, 0, stream>>>(out, hinv, HCH);
    bn_stats_kernel<<<ROWS / 32, 256, 0, stream>>>(out, hinv, sums, ssums);
    bn_apply_kernel<<<8192, 256, 0, stream>>>(out, sums, ssums, gamma, beta);
}
